// PanoformerBlock_81578608820347
// MI455X (gfx1250) — matrix operand order repaired and checked by reading, not compiled, not run on hardware
//
#include <hip/hip_runtime.h>
#include <hip/hip_bf16.h>
#include <math.h>

// ---------------- problem constants ----------------
#define BB   2
#define HH   64
#define WW   128
#define CC   256
#define NHH  8
#define KK   9
#define HD   32           // CC / NHH == wave32 width (lane == channel-in-head)
#define HID  1024
#define NN   (HH*WW)      // 8192
#define MN   (BB*NN)      // 16384 rows for every GEMM (divisible by 512)
#define NOFF 144          // NH*K*2 (multiple of 16)
#define NATT 72           // NH*K
#define NATTP 80          // padded to 16
#define KMAX 1024         // largest GEMM K (LeFF w2)

typedef _Float16 h16_t;
typedef _Float16 v16h __attribute__((ext_vector_type(16)));
typedef _Float16 h8   __attribute__((ext_vector_type(8)));
typedef float    v8f  __attribute__((ext_vector_type(8)));
typedef int      v4i  __attribute__((ext_vector_type(4)));

typedef __attribute__((address_space(1))) v4i* gv4i_p;   // global-AS v4i*
typedef __attribute__((address_space(3))) v4i* sv4i_p;   // LDS-AS v4i*

union frag16 { v16h v; h8 h[2]; };

#if defined(__has_builtin)
#if __has_builtin(__builtin_amdgcn_global_load_async_to_lds_b128)
#define HAVE_ASYNC_LDS 1
#endif
#endif

__device__ __forceinline__ float gelu_exact(float x) {
    return 0.5f * x * (1.0f + erff(x * 0.70710678118654752f));
}

// ---------------- LayerNorm -> f16 ----------------
__global__ __launch_bounds__(CC) void k_layernorm_f16(
    const float* __restrict__ x, const float* __restrict__ g,
    const float* __restrict__ b, h16_t* __restrict__ out) {
    __shared__ float s1[CC], s2[CC];
    const int row = blockIdx.x;
    const int c   = threadIdx.x;
    const float v = x[(long)row * CC + c];
    s1[c] = v; s2[c] = v * v;
    __syncthreads();
    for (int off = CC / 2; off > 0; off >>= 1) {
        if (c < off) { s1[c] += s1[c + off]; s2[c] += s2[c + off]; }
        __syncthreads();
    }
    const float mean = s1[0] * (1.0f / CC);
    const float var  = s2[0] * (1.0f / CC) - mean * mean;
    const float inv  = rsqrtf(var + 1e-5f);
    out[(long)row * CC + c] = (h16_t)((v - mean) * inv * g[c] + b[c]);
}

// ---------------- weight f32 (K x Nsrc) -> f16 TRANSPOSED (Npad x K) ----------------
__global__ void k_w_to_f16t(const float* __restrict__ src, h16_t* __restrict__ dst,
                            int Kd, int Nsrc, int Npad) {
    const int i = blockIdx.x * blockDim.x + threadIdx.x;
    if (i >= Kd * Npad) return;
    const int n = i / Kd, k = i % Kd;
    dst[i] = (n < Nsrc) ? (h16_t)src[(long)k * Nsrc + n] : (h16_t)0.0f;
}

// ---------------- WMMA GEMM: LDS-staged B, double-buffered, fused epilogue ----------------
// C[M x Np] = A[M x Kd] (f16 row-major) * Bt[Np x Kd]^T (f16) + bias (+gelu)(+res)
// Block = 8 waves = one N-tile (16 cols) x 512 rows. The 16 x Kd B tile is
// async-copied to LDS once (ASYNCcnt); B fragments come from LDS via
// ds_load_b128 pairs, A streams from global (L2-resident across N-tiles).
// The K loop is software-pipelined: fragment loads for slab i+1 issue before
// the 4 WMMAs of slab i, so each load batch hides under a full WMMA group.
// Requires Kd % 64 == 0 (true for 256 and 1024).
// Fragment layouts per CDNA5 ISA 7.12.2 (wave32):
//   A 16x32 f16 : M = lane&15, K(e) = (e&7)+16*(e>>3)+8*(lane>>4)
//   B 32x16 f16 : N = lane&15, K(e) = (e&7)+16*(e>>3)+8*(lane>>4)
//   D 16x16 f32 : N = lane&15, M(g) = g + 8*(lane>>4)
__global__ __launch_bounds__(256) void k_gemm_wmma(
    const h16_t* __restrict__ A, const h16_t* __restrict__ Bt,
    const float* __restrict__ bias, int nbias,
    const float* __restrict__ res,        // nullable residual, M x Np
    float*  __restrict__ outF,            // nullable f32 out
    h16_t*  __restrict__ outH,            // nullable f16 out
    int M, int Kd, int Np, int act_gelu) {
    __shared__ h16_t Bs[16 * KMAX];                // <= 32 KB of the 320 KB WGP LDS

    const int lane = threadIdx.x & 31;
    const int wave = threadIdx.x >> 5;
    const int groupsM = M >> 9;                    // 512-row groups
    const int gm = blockIdx.x % groupsM;
    const int tn = blockIdx.x / groupsM;

    // ---- stage B tile (16 x Kd halfs) into LDS via async-to-LDS ----
    const h16_t* __restrict__ gB = Bt + (long)(tn * 16) * Kd;
    const int chunks = (16 * Kd) / 8;              // 16-byte chunks
    for (int i = threadIdx.x; i < chunks; i += 256) {
#if defined(HAVE_ASYNC_LDS)
        __builtin_amdgcn_global_load_async_to_lds_b128(
            (gv4i_p)(gB + i * 8), (sv4i_p)(&Bs[i * 8]), 0, 0);
#else
        *(h8*)&Bs[i * 8] = *(const h8*)(gB + i * 8);
#endif
    }
#if defined(HAVE_ASYNC_LDS)
#if __has_builtin(__builtin_amdgcn_s_wait_asynccnt)
    __builtin_amdgcn_s_wait_asynccnt(0);
#else
    asm volatile("s_wait_asynccnt 0x0" ::: "memory");
#endif
#endif
    __syncthreads();

    const int r  = lane & 15;
    const int hi = lane >> 4;
    const int n  = tn * 16 + r;
    const int row0 = gm * 512 + wave * 64 + r;     // wave covers 64 rows (4 subtiles)
    const h16_t* __restrict__ Arow0 = A + (long)row0 * Kd;
    const h16_t* __restrict__ Bsn   = Bs + r * Kd; // lane's column, K-contiguous

    v8f acc[4];
    #pragma unroll
    for (int t = 0; t < 4; ++t) acc[t] = (v8f){0.f,0.f,0.f,0.f,0.f,0.f,0.f,0.f};

    auto loadA = [&](frag16* af, int kk) {
        #pragma unroll
        for (int t = 0; t < 4; ++t) {
            const h16_t* Ar = Arow0 + (long)(t * 16) * Kd;
            af[t].h[0] = *(const h8*)(Ar + kk + 8 * hi);
            af[t].h[1] = *(const h8*)(Ar + kk + 16 + 8 * hi);
        }
    };
    auto loadB = [&](frag16& bf, int kk) {
        bf.h[0] = *(const h8*)(Bsn + kk + 8 * hi);
        bf.h[1] = *(const h8*)(Bsn + kk + 16 + 8 * hi);
    };
    auto mma4 = [&](frag16* af, frag16& bf) {
        #pragma unroll
        for (int t = 0; t < 4; ++t)
            acc[t] = __builtin_amdgcn_wmma_f32_16x16x32_f16(
                         false, af[t].v, false, bf.v, (short)0, acc[t], false, false);
    };

    frag16 afA[4], afB[4], bfA, bfB;
    loadB(bfA, 0);
    loadA(afA, 0);
    for (int k0 = 0; k0 < Kd; k0 += 64) {          // two slabs per iteration
        if (k0 + 64 < Kd)                          // next A slab -> global_prefetch_b8
            __builtin_prefetch(Arow0 + k0 + 64, 0, 3);
        loadB(bfB, k0 + 32);                       // slab i+1 loads in flight...
        loadA(afB, k0 + 32);
        mma4(afA, bfA);                            // ...under slab i's 4 WMMAs
        if (k0 + 64 < Kd) {
            loadB(bfA, k0 + 64);
            loadA(afA, k0 + 64);
        }
        mma4(afB, bfB);
    }

    const float bn = (bias && n < nbias) ? bias[n] : 0.0f;
    #pragma unroll
    for (int t = 0; t < 4; ++t) {
        #pragma unroll
        for (int g = 0; g < 8; ++g) {
            const int  mrow = gm * 512 + wave * 64 + t * 16 + g + 8 * hi;
            const long o    = (long)mrow * Np + n;
            float v = acc[t][g] + bn;
            if (act_gelu) v = gelu_exact(v);
            if (res)  v += res[o];
            if (outF) outF[o] = v;
            if (outH) outH[o] = (h16_t)v;
        }
    }
}

// ---------------- deformable sampling ----------------
// one wave per (b, n, head); lane == channel d (HD == 32 == wave32)
__global__ __launch_bounds__(256) void k_deform(
    const float* __restrict__ v,    // (B,N,C) value projection
    const float* __restrict__ off,  // (B,N,144)
    const float* __restrict__ att,  // (B,N,80) logits (cols >= 72 unused)
    const float* __restrict__ ref,  // (N,K,2) in [0,1]
    h16_t* __restrict__ ao) {       // (B,N,C) f16 attention output
    const int h    = threadIdx.x >> 5;
    const int lane = threadIdx.x & 31;
    const int bn   = blockIdx.x;
    const int b = bn / NN, n = bn % NN;

    const float* lo = off + (long)bn * NOFF  + h * KK * 2;
    const float* la = att + (long)bn * NATTP + h * KK;
    const float* lr = ref + (long)n  * KK * 2;

    float aw[KK];
    float mx = -1e30f;
    #pragma unroll
    for (int k = 0; k < KK; ++k) mx = fmaxf(mx, la[k]);
    float s = 0.f;
    #pragma unroll
    for (int k = 0; k < KK; ++k) { aw[k] = expf(la[k] - mx); s += aw[k]; }
    const float invs = 1.0f / s;

    const float* vb = v + (long)b * NN * CC + h * HD + lane;
    float acc = 0.f;
    #pragma unroll
    for (int k = 0; k < KK; ++k) {
        float px = lr[2 * k + 0] * (float)(WW - 1) + lo[2 * k + 0];
        float py = lr[2 * k + 1] * (float)(HH - 1) + lo[2 * k + 1];
        px = fminf(fmaxf(px, 0.f), (float)(WW - 1));
        py = fminf(fmaxf(py, 0.f), (float)(HH - 1));
        const float x0f = floorf(px), y0f = floorf(py);
        const int x0 = (int)x0f, y0 = (int)y0f;
        const int x1 = (x0 + 1 < WW) ? x0 + 1 : WW - 1;
        const int y1 = (y0 + 1 < HH) ? y0 + 1 : HH - 1;
        const float wx = px - x0f, wy = py - y0f;
        const float g00 = vb[(long)(y0 * WW + x0) * CC];   // coalesced over lanes
        const float g01 = vb[(long)(y0 * WW + x1) * CC];
        const float g10 = vb[(long)(y1 * WW + x0) * CC];
        const float g11 = vb[(long)(y1 * WW + x1) * CC];
        const float smp = g00 * (1 - wx) * (1 - wy) + g01 * wx * (1 - wy)
                        + g10 * (1 - wx) * wy       + g11 * wx * wy;
        acc += aw[k] * invs * smp;
    }
    ao[(long)bn * CC + h * HD + lane] = (h16_t)acc;
}

// ---------------- depthwise 3x3 conv (circular W, zero H) + GELU ----------------
__global__ __launch_bounds__(256) void k_dwconv(
    const h16_t* __restrict__ hin,   // (B,N,HID) f16
    const float* __restrict__ wdw,   // (HID,1,3,3)
    const float* __restrict__ bdw,   // (HID,)
    h16_t* __restrict__ hout) {      // (B,N,HID) f16
    const int c   = (blockIdx.x & 3) * 256 + threadIdx.x;
    const int pix = blockIdx.x >> 2;
    const int b = pix / NN, n = pix % NN;
    const int y = n / WW,  x = n % WW;
    float acc = 0.f;
    #pragma unroll
    for (int dy = -1; dy <= 1; ++dy) {
        const int yy = y + dy;
        if (yy < 0 || yy >= HH) continue;          // zero pad along H
        #pragma unroll
        for (int dx = -1; dx <= 1; ++dx) {
            const int xx = (x + dx + WW) % WW;     // circular pad along W
            acc += (float)hin[((long)b * NN + yy * WW + xx) * HID + c]
                 * wdw[c * 9 + (dy + 1) * 3 + (dx + 1)];
        }
    }
    hout[((long)b * NN + n) * HID + c] = (h16_t)gelu_exact(acc + bdw[c]);
}

// ---------------- host orchestration ----------------
static inline void launch_gemm(const h16_t* A, const h16_t* Bt, const float* bias,
                               int nbias, const float* res, float* outF, h16_t* outH,
                               int M, int Kd, int Np, int act, hipStream_t s) {
    const int blocks = (M >> 9) * (Np >> 4);       // 512-row x 16-col tile per block
    k_gemm_wmma<<<blocks, 256, 0, s>>>(A, Bt, bias, nbias, res, outF, outH, M, Kd, Np, act);
}

extern "C" void kernel_launch(void* const* d_in, const int* in_sizes, int n_in,
                              void* d_out, int out_size, void* d_ws, size_t ws_size,
                              hipStream_t stream) {
    (void)in_sizes; (void)n_in; (void)out_size; (void)ws_size;
    const float* x    = (const float*)d_in[0];
    const float* ref  = (const float*)d_in[1];
    const float* n1g  = (const float*)d_in[2];
    const float* n1b  = (const float*)d_in[3];
    const float* w_v  = (const float*)d_in[4];
    const float* b_v  = (const float*)d_in[5];
    const float* w_o  = (const float*)d_in[6];
    const float* b_o  = (const float*)d_in[7];
    const float* w_a  = (const float*)d_in[8];
    const float* b_a  = (const float*)d_in[9];
    const float* w_out= (const float*)d_in[10];
    const float* b_out= (const float*)d_in[11];
    const float* n2g  = (const float*)d_in[12];
    const float* n2b  = (const float*)d_in[13];
    const float* w1   = (const float*)d_in[14];
    const float* b1   = (const float*)d_in[15];
    const float* w_dw = (const float*)d_in[16];
    const float* b_dw = (const float*)d_in[17];
    const float* w2   = (const float*)d_in[18];
    const float* b2   = (const float*)d_in[19];
    float* out = (float*)d_out;

    // -------- workspace carve-up --------
    char* ws = (char*)d_ws;
    size_t cur = 0;
    auto alloc = [&](size_t bytes) -> void* {
        void* p = ws + cur;
        cur += (bytes + 255) & ~(size_t)255;
        return p;
    };
    h16_t* xn16   = (h16_t*)alloc((size_t)MN * CC   * 2);  // LN output (reused LN1/LN2)
    h16_t* ao16   = (h16_t*)alloc((size_t)MN * CC   * 2);  // attention output f16
    float* vbuf   = (float*)alloc((size_t)MN * CC   * 4);  // value projection
    float* offb   = (float*)alloc((size_t)MN * NOFF * 4);  // offsets
    float* attb   = (float*)alloc((size_t)MN * NATTP* 4);  // attn logits (padded)
    float* xres   = (float*)alloc((size_t)MN * CC   * 4);  // x + attn residual
    h16_t* h16buf = (h16_t*)alloc((size_t)MN * HID  * 2);  // gelu(xn2@w1+b1)
    h16_t* hdw16  = (h16_t*)alloc((size_t)MN * HID  * 2);  // gelu(dwconv)
    h16_t* wv16   = (h16_t*)alloc((size_t)CC * CC    * 2); // all weights stored N x K (f16)
    h16_t* woff16 = (h16_t*)alloc((size_t)NOFF * CC  * 2);
    h16_t* wat16  = (h16_t*)alloc((size_t)NATTP * CC * 2);
    h16_t* wo16   = (h16_t*)alloc((size_t)CC * CC    * 2);
    h16_t* w116   = (h16_t*)alloc((size_t)HID * CC   * 2);
    h16_t* w216   = (h16_t*)alloc((size_t)CC * HID   * 2);

    // -------- weights -> f16, transposed (Npad x K), zero-padded --------
    auto cvt = [&](const float* s, h16_t* d, int Kd, int Ns, int Npad) {
        const int tot = Kd * Npad;
        k_w_to_f16t<<<(tot + 255) / 256, 256, 0, stream>>>(s, d, Kd, Ns, Npad);
    };
    cvt(w_v,  wv16,   CC,  CC,   CC);
    cvt(w_o,  woff16, CC,  NOFF, NOFF);
    cvt(w_a,  wat16,  CC,  NATT, NATTP);
    cvt(w_out,wo16,   CC,  CC,   CC);
    cvt(w1,   w116,   CC,  HID,  HID);
    cvt(w2,   w216,   HID, CC,   CC);

    // -------- LN1 --------
    k_layernorm_f16<<<MN, CC, 0, stream>>>(x, n1g, n1b, xn16);

    // -------- value / offset / attn projections (WMMA) --------
    launch_gemm(xn16, wv16,   b_v, CC,   nullptr, vbuf, nullptr, MN, CC, CC,    0, stream);
    launch_gemm(xn16, woff16, b_o, NOFF, nullptr, offb, nullptr, MN, CC, NOFF,  0, stream);
    launch_gemm(xn16, wat16,  b_a, NATT, nullptr, attb, nullptr, MN, CC, NATTP, 0, stream);

    // -------- deformable sampling (wave per (b,n,head)) --------
    k_deform<<<MN, 256, 0, stream>>>(vbuf, offb, attb, ref, ao16);

    // -------- out projection + residual (WMMA, fused) --------
    launch_gemm(ao16, wo16, b_out, CC, x, xres, nullptr, MN, CC, CC, 0, stream);

    // -------- LN2 --------
    k_layernorm_f16<<<MN, CC, 0, stream>>>(xres, n2g, n2b, xn16);

    // -------- LeFF w1 + GELU (WMMA, f16 out) --------
    launch_gemm(xn16, w116, b1, HID, nullptr, nullptr, h16buf, MN, CC, HID, 1, stream);

    // -------- depthwise conv + GELU --------
    k_dwconv<<<MN * 4, 256, 0, stream>>>(h16buf, w_dw, b_dw, hdw16);

    // -------- LeFF w2 + residual -> d_out (WMMA, fused) --------
    launch_gemm(hdw16, w216, b2, CC, xres, out, nullptr, MN, HID, CC, 0, stream);
}
